// Model_45200235823595
// MI455X (gfx1250) — matrix operand order repaired and checked by reading, not compiled, not run on hardware
//
#include <hip/hip_runtime.h>

// ---------------------------------------------------------------------------
// Types for CDNA5 WMMA (wave32, v_wmma_f32_16x16x32_bf16)
// ---------------------------------------------------------------------------
typedef __bf16 bf16;
typedef __attribute__((ext_vector_type(16))) __bf16 v16bf;
typedef __attribute__((ext_vector_type(8)))  float  v8f;
typedef __attribute__((ext_vector_type(4)))  unsigned int v4u;

union ABf { v16bf v; v4u q[2]; };
union DFf { v8f  v; float f[8]; };

#define MODE_NORM 0   // 3x3x3, stride1, SAME
#define MODE_DOWN 1   // 2x2x2, stride2 (in = 2*out + k)
#define MODE_UP   2   // 2x2x2 transpose stride2 (out = 2*in + k)
#define MODE_NIN  3   // 1x1x1

// ---------------------------------------------------------------------------
// Weight re-pack: f32 [T][Cin][Cout] -> bf16 B-fragment blocks.
// Block (t, kc, coT) = 32 lanes x 8 dwords; lane l: n=l&15, h=l>>4,
// dword r holds bf16 pair (K = kc*32 + 2r + 8h + 8*(r>=4), K+1) for column co = coT*16+n.
// ---------------------------------------------------------------------------
__global__ __launch_bounds__(256) void k_pack(const float* __restrict__ w,
                                              bf16* __restrict__ wpk,
                                              int T, int Cin, int Cout)
{
  long gid = (long)blockIdx.x * 256 + threadIdx.x;
  long total = (long)T * Cin * Cout / 2;       // one thread per bf16 pair
  if (gid >= total) return;
  int r    = (int)(gid & 7);
  int lane = (int)((gid >> 3) & 31);
  long blk = gid >> 8;
  int nCT = Cout >> 4;
  int KC  = Cin >> 5;
  int coT = (int)(blk % nCT);
  int kT  = (int)((blk / nCT) % KC);
  int t   = (int)(blk / ((long)nCT * KC));
  int hf = lane >> 4, n = lane & 15;
  int k0 = kT * 32 + 2 * r + hf * 8 + ((r >> 2) << 3);
  int co = coT * 16 + n;
  float va = w[((long)t * Cin + k0)     * Cout + co];
  float vb = w[((long)t * Cin + k0 + 1) * Cout + co];
  bf16* dst = wpk + blk * 512 + lane * 16 + r * 2;
  dst[0] = (bf16)va;
  dst[1] = (bf16)vb;
}

// ---------------------------------------------------------------------------
// Implicit-GEMM masked submanifold conv via WMMA.
// One wave computes a 32-voxel x 32-Cout macro-tile:
//   2 voxel tiles (A0, A1) x 2 Cout tiles (B0, B1) -> 4 accumulators.
// A fragments reused across Cout tiles, B across voxel tiles:
// 2 b128 loads per WMMA, 4 independent WMMA chains per K-step.
// Boundary taps redirect the A-row pointer to a zeroed ghost row (zrow),
// so all loads are unconditional (no EXEC toggling in the hot loop).
// ---------------------------------------------------------------------------
__global__ __launch_bounds__(256) void k_conv(const bf16* __restrict__ x,
                                              const bf16* __restrict__ wpk,
                                              const bf16* __restrict__ zrow,
                                              float* __restrict__ y,
                                              const float* __restrict__ omask,
                                              int mode, int Sin, int Sout,
                                              int Cin, int Cout, int nTaps)
{
  const int lane = threadIdx.x & 31;
  const int wv   = (int)(blockIdx.x * 8u + (threadIdx.x >> 5));
  const int nCT  = Cout >> 4;
  const int nCT2 = nCT >> 1;                 // Cout is always a multiple of 32
  const long Vout = (long)Sout * Sout * Sout;
  const int nVT2 = (int)((Vout + 31) >> 5);
  if (wv >= nVT2 * nCT2) return;             // whole-wave uniform exit

  const int coT2 = wv % nCT2;
  const int vT2  = wv / nCT2;
  const int co0  = coT2 << 5;                // first of 32 output channels
  const int hf   = lane >> 4;                // wave32 half
  const int mrow = lane & 15;                // A-fragment row within tile
  const long vbase = (long)vT2 * 32;

  // Per-lane output rows for the two voxel tiles.
  const long vo0 = vbase + mrow;
  const long vo1 = vbase + 16 + mrow;
  const bool rOk0 = vo0 < Vout;
  const bool rOk1 = vo1 < Vout;
  int xo0=0, yo0=0, zo0=0, xo1=0, yo1=0, zo1=0;
  if (rOk0) { zo0=(int)(vo0%Sout); yo0=(int)((vo0/Sout)%Sout); xo0=(int)(vo0/((long)Sout*Sout)); }
  if (rOk1) { zo1=(int)(vo1%Sout); yo1=(int)((vo1/Sout)%Sout); xo1=(int)(vo1/((long)Sout*Sout)); }

  v8f acc00 = {0.f,0.f,0.f,0.f,0.f,0.f,0.f,0.f};
  v8f acc01 = acc00, acc10 = acc00, acc11 = acc00;
  const int KC = Cin >> 5;

  for (int t = 0; t < nTaps; ++t) {
    long vi0 = 0, vi1 = 0; bool ok0 = rOk0, ok1 = rOk1;
    if (mode == MODE_NORM) {
      int dx = t / 9 - 1, dy = (t / 3) % 3 - 1, dz = t % 3 - 1;
      int a = xo0+dx, b = yo0+dy, c = zo0+dz;
      ok0 = ok0 && (unsigned)a < (unsigned)Sin && (unsigned)b < (unsigned)Sin
                && (unsigned)c < (unsigned)Sin;
      vi0 = ((long)a * Sin + b) * Sin + c;
      a = xo1+dx; b = yo1+dy; c = zo1+dz;
      ok1 = ok1 && (unsigned)a < (unsigned)Sin && (unsigned)b < (unsigned)Sin
                && (unsigned)c < (unsigned)Sin;
      vi1 = ((long)a * Sin + b) * Sin + c;
    } else if (mode == MODE_DOWN) {
      int dx = (t >> 2) & 1, dy = (t >> 1) & 1, dz = t & 1;
      vi0 = ((long)(2*xo0+dx) * Sin + (2*yo0+dy)) * Sin + (2*zo0+dz);
      vi1 = ((long)(2*xo1+dx) * Sin + (2*yo1+dy)) * Sin + (2*zo1+dz);
    } else if (mode == MODE_UP) {
      int k0t = ((xo0&1)<<2) | ((yo0&1)<<1) | (zo0&1);
      int k1t = ((xo1&1)<<2) | ((yo1&1)<<1) | (zo1&1);
      ok0 = ok0 && (t == k0t);
      ok1 = ok1 && (t == k1t);
      vi0 = ((long)(xo0>>1) * Sin + (yo0>>1)) * Sin + (zo0>>1);
      vi1 = ((long)(xo1>>1) * Sin + (yo1>>1)) * Sin + (zo1>>1);
    } else {                                // MODE_NIN
      vi0 = vo0; vi1 = vo1;
    }

    // Pointer-select boundary handling: invalid rows read the zero ghost row.
    const bf16* arow0 = ok0 ? (x + vi0 * Cin) : zrow;
    const bf16* arow1 = ok1 ? (x + vi1 * Cin) : zrow;
    const bf16* bbase = wpk + ((long)(t * KC) * nCT + (coT2 << 1)) * 512;
    __builtin_prefetch(bbase + (long)KC * nCT * 512, 0, 1);  // next tap's B

    for (int kc = 0; kc < KC; ++kc) {
      // A fragments: ISA 16-bit A 16x32 layout; lane half h covers
      // K in [kc*32+8h, +8) and [kc*32+16+8h, +8) -> two b128 loads each.
      const int aoff = (kc << 5) + (hf << 3);
      ABf a0, a1;
      a0.q[0] = *(const v4u*)(arow0 + aoff);
      a0.q[1] = *(const v4u*)(arow0 + aoff + 16);
      a1.q[0] = *(const v4u*)(arow1 + aoff);
      a1.q[1] = *(const v4u*)(arow1 + aoff + 16);
      // B fragments: pre-packed, 32 contiguous bytes per lane; the two
      // Cout tiles are adjacent 1KB blocks.
      const bf16* bp = bbase + (long)kc * nCT * 512 + lane * 16;
      ABf b0, b1;
      b0.q[0] = *(const v4u*)bp;
      b0.q[1] = *(const v4u*)(bp + 8);
      b1.q[0] = *(const v4u*)(bp + 512);
      b1.q[1] = *(const v4u*)(bp + 520);

      acc00 = __builtin_amdgcn_wmma_f32_16x16x32_bf16(false, a0.v, false, b0.v, (short)0, acc00, false, false);
      acc01 = __builtin_amdgcn_wmma_f32_16x16x32_bf16(false, a0.v, false, b1.v, (short)0, acc01, false, false);
      acc10 = __builtin_amdgcn_wmma_f32_16x16x32_bf16(false, a1.v, false, b0.v, (short)0, acc10, false, false);
      acc11 = __builtin_amdgcn_wmma_f32_16x16x32_bf16(false, a1.v, false, b1.v, (short)0, acc11, false, false);
    }
  }

  // D layout: VGPR r -> row m = r + 8*half, col n = lane&15.
  DFf d00, d01, d10, d11;
  d00.v = acc00; d01.v = acc01; d10.v = acc10; d11.v = acc11;
  const int n = lane & 15;
  for (int r = 0; r < 8; ++r) {
    long vv0 = vbase + r + (hf << 3);
    long vv1 = vv0 + 16;
    if (vv0 < Vout) {
      float mk = omask[vv0];
      float* yp = y + vv0 * Cout + co0 + n;
      yp[0]  = d00.f[r] * mk;
      yp[16] = d01.f[r] * mk;
    }
    if (vv1 < Vout) {
      float mk = omask[vv1];
      float* yp = y + vv1 * Cout + co0 + n;
      yp[0]  = d10.f[r] * mk;
      yp[16] = d11.f[r] * mk;
    }
  }
}

// ---------------------------------------------------------------------------
// BatchNorm over active sites (x is pre-masked so plain sums == active sums)
// ---------------------------------------------------------------------------
__global__ __launch_bounds__(256) void k_bnstats(const float* __restrict__ x,
                                                 float* __restrict__ st,
                                                 long V, int C)
{
  long v0 = (long)blockIdx.x * 64;
  long v1 = v0 + 64; if (v1 > V) v1 = V;
  for (int c = threadIdx.x; c < C; c += 256) {
    float s = 0.f, ss = 0.f;
    for (long v = v0; v < v1; ++v) {
      float f = x[v * C + c];
      s += f; ss += f * f;
    }
    atomicAdd(&st[c], s);
    atomicAdd(&st[C + c], ss);
  }
}

__global__ void k_bnfin(float* st, const float* __restrict__ g,
                        const float* __restrict__ b, const float* __restrict__ cnt,
                        int C, float eps)
{
  int c = blockIdx.x * blockDim.x + threadIdx.x;
  if (c >= C) return;
  float nactive = fmaxf(cnt[0], 1.f);
  float mean = st[c] / nactive;
  float var  = st[C + c] / nactive - mean * mean;
  float sc   = g[c] * rsqrtf(fmaxf(var, 0.f) + eps);
  float sh   = b[c] - mean * sc;
  st[c] = sc; st[C + c] = sh;
}

__global__ __launch_bounds__(256) void k_bnapply(const float* __restrict__ x,
                                                 const float* __restrict__ st,
                                                 const float* __restrict__ mask,
                                                 bf16* __restrict__ o,
                                                 long V, int C)
{
  long i = (long)blockIdx.x * 256 + threadIdx.x;
  if (i >= V * C) return;
  long v = i / C; int c = (int)(i - v * C);
  float val = fmaxf(x[i] * st[c] + st[C + c], 0.f) * mask[v];
  o[i] = (bf16)val;
}

__global__ __launch_bounds__(256) void k_count(const float* __restrict__ m,
                                               float* __restrict__ cnt, long V)
{
  __shared__ float s[256];
  float acc = 0.f;
  for (long j = (long)blockIdx.x * 256 + threadIdx.x; j < V;
       j += (long)gridDim.x * 256) acc += m[j];
  s[threadIdx.x] = acc; __syncthreads();
  for (int o = 128; o > 0; o >>= 1) {
    if ((int)threadIdx.x < o) s[threadIdx.x] += s[threadIdx.x + o];
    __syncthreads();
  }
  if (threadIdx.x == 0) atomicAdd(cnt, s[0]);
}

// ---------------------------------------------------------------------------
// Elementwise helpers
// ---------------------------------------------------------------------------
__global__ __launch_bounds__(256) void k_add(const float* __restrict__ a,
                                             const float* __restrict__ b,
                                             float* __restrict__ o, long n)
{
  long i = (long)blockIdx.x * 256 + threadIdx.x;
  if (i < n) o[i] = a[i] + b[i];
}

__global__ __launch_bounds__(256) void k_tobf16(const float* __restrict__ x,
                                                bf16* __restrict__ o, long n)
{
  long i = (long)blockIdx.x * 256 + threadIdx.x;
  if (i < n) o[i] = (bf16)x[i];
}

__global__ __launch_bounds__(256) void k_concat(const float* __restrict__ a,
                                                const float* __restrict__ b,
                                                float* __restrict__ o,
                                                long V, int C)
{
  long i = (long)blockIdx.x * 256 + threadIdx.x;
  long n = V * 2 * C;
  if (i >= n) return;
  long v = i / (2 * C); int c = (int)(i - v * 2 * C);
  o[i] = (c < C) ? a[v * C + c] : b[v * C + (c - C)];
}

// ---------------------------------------------------------------------------
// Input layer: scatter-average points to grid; build mask
// ---------------------------------------------------------------------------
__global__ __launch_bounds__(256) void k_scatter(const float* __restrict__ feats,
                                                 const int* __restrict__ coords,
                                                 float* __restrict__ sums,
                                                 float* __restrict__ cnt,
                                                 int N, int S)
{
  int p = blockIdx.x * 256 + threadIdx.x;
  if (p >= N) return;
  const int* cd = coords + (long)p * 4;
  long lin = (((long)cd[3] * S + cd[0]) * S + cd[1]) * S + cd[2];
  atomicAdd(&sums[lin * 3 + 0], feats[(long)p * 3 + 0]);
  atomicAdd(&sums[lin * 3 + 1], feats[(long)p * 3 + 1]);
  atomicAdd(&sums[lin * 3 + 2], feats[(long)p * 3 + 2]);
  atomicAdd(&cnt[lin], 1.f);
}

__global__ __launch_bounds__(256) void k_norm(const float* __restrict__ sums,
                                              const float* __restrict__ cnt,
                                              float* __restrict__ grid,
                                              float* __restrict__ mask, long V)
{
  long v = (long)blockIdx.x * 256 + threadIdx.x;
  if (v >= V) return;
  float c = cnt[v];
  float d = fmaxf(c, 1.f);
  grid[v * 3 + 0] = sums[v * 3 + 0] / d;
  grid[v * 3 + 1] = sums[v * 3 + 1] / d;
  grid[v * 3 + 2] = sums[v * 3 + 2] / d;
  mask[v] = (c > 0.f) ? 1.f : 0.f;
}

// Stem conv: Cin=3 (not WMMA-friendly) -> scalar; negligible FLOPs.
__global__ __launch_bounds__(256) void k_stem(const float* __restrict__ grid,
                                              const float* __restrict__ mask,
                                              const float* __restrict__ w,
                                              float* __restrict__ y, int S)
{
  long V = (long)S * S * S;
  long i = (long)blockIdx.x * 256 + threadIdx.x;
  if (i >= V * 32) return;
  long v = i >> 5; int co = (int)(i & 31);
  int z = (int)(v % S), yy = (int)((v / S) % S), x = (int)(v / ((long)S * S));
  float acc = 0.f;
  for (int t = 0; t < 27; ++t) {
    int dx = t / 9 - 1, dy = (t / 3) % 3 - 1, dz = t % 3 - 1;
    int xi = x + dx, yi = yy + dy, zi = z + dz;
    if ((unsigned)xi >= (unsigned)S || (unsigned)yi >= (unsigned)S ||
        (unsigned)zi >= (unsigned)S) continue;
    long vi = ((long)xi * S + yi) * S + zi;
    for (int ci = 0; ci < 3; ++ci)
      acc += grid[vi * 3 + ci] * w[((long)t * 3 + ci) * 32 + co];
  }
  y[i] = acc * mask[v];
}

__global__ __launch_bounds__(256) void k_pool(const float* __restrict__ mi,
                                              float* __restrict__ mo, int So)
{
  long Vo = (long)So * So * So;
  long i = (long)blockIdx.x * 256 + threadIdx.x;
  if (i >= Vo) return;
  int Si = 2 * So;
  int z = (int)(i % So), y = (int)((i / So) % So), x = (int)(i / ((long)So * So));
  float m = 0.f;
  for (int d = 0; d < 8; ++d) {
    int dx = (d >> 2) & 1, dy = (d >> 1) & 1, dz = d & 1;
    m = fmaxf(m, mi[(((long)(2 * x + dx)) * Si + (2 * y + dy)) * Si + (2 * z + dz)]);
  }
  mo[i] = m;
}

// Output layer: gather per-point bf16 features, 32->20 linear.
__global__ __launch_bounds__(256) void k_final(const bf16* __restrict__ f,
                                               const int* __restrict__ coords,
                                               const float* __restrict__ lw,
                                               const float* __restrict__ lb,
                                               float* __restrict__ out,
                                               int N, int S)
{
  long i = (long)blockIdx.x * 256 + threadIdx.x;
  if (i >= (long)N * 20) return;
  int p = (int)(i / 20), c = (int)(i - (long)p * 20);
  const int* cd = coords + (long)p * 4;
  long lin = (((long)cd[3] * S + cd[0]) * S + cd[1]) * S + cd[2];
  float acc = lb[c];
  for (int k = 0; k < 32; ++k)
    acc += (float)f[lin * 32 + k] * lw[(long)k * 20 + c];
  out[i] = acc;
}

// ---------------------------------------------------------------------------
// Host-side orchestration
// ---------------------------------------------------------------------------
struct Lv {
  int S; long V; int C;
  float *xa, *xb, *y1, *y2, *xu, *sc, *cat, *mask, *cnt;
  bf16 *tbf, *catbf;
};

struct Ctx {
  void* const* din;
  hipStream_t st;
  char* base; size_t off;
  float* stats;
  bf16* zrow;            // 512 zeroed bf16 (ghost row for boundary taps)
  int pi;
  Lv lv[7];
};

static void* ws_alloc(Ctx& c, size_t bytes) {
  void* p = c.base + c.off;
  c.off = (c.off + bytes + 255) & ~(size_t)255;
  return p;
}

static inline const float* nextP(Ctx& cx) {
  return (const float*)cx.din[cx.pi++];
}

static void do_bnrelu(Ctx& cx, const float* x, const Lv& L, int C,
                      const float* g, const float* b, bf16* o)
{
  hipMemsetAsync(cx.stats, 0, (size_t)2 * C * sizeof(float), cx.st);
  unsigned nb = (unsigned)((L.V + 63) / 64);
  k_bnstats<<<nb, 256, 0, cx.st>>>(x, cx.stats, L.V, C);
  k_bnfin<<<(unsigned)((C + 63) / 64), 64, 0, cx.st>>>(cx.stats, g, b, L.cnt, C, 1e-4f);
  long tot = L.V * C;
  k_bnapply<<<(unsigned)((tot + 255) / 256), 256, 0, cx.st>>>(x, cx.stats, L.mask, o, L.V, C);
}

static void do_conv(Ctx& cx, const bf16* xin, const float* wf, int T,
                    int Cin, int Cout, float* yout, const float* omask,
                    int mode, int Sin, int Sout)
{
  long pkBytes = (long)T * Cin * Cout * 2;
  bf16* wpk = (bf16*)ws_alloc(cx, (size_t)pkBytes);
  long pairs = (long)T * Cin * Cout / 2;
  k_pack<<<(unsigned)((pairs + 255) / 256), 256, 0, cx.st>>>(wf, wpk, T, Cin, Cout);
  long Vout = (long)Sout * Sout * Sout;
  long waves = ((Vout + 31) / 32) * (Cout / 32);
  k_conv<<<(unsigned)((waves + 7) / 8), 256, 0, cx.st>>>(
      xin, wpk, cx.zrow, yout, omask, mode, Sin, Sout, Cin, Cout, T);
}

static void do_resblock(Ctx& cx, Lv& L, const float* xin, float* xout, int Cin)
{
  const float* g1 = nextP(cx); const float* b1 = nextP(cx); const float* w1 = nextP(cx);
  const float* g2 = nextP(cx); const float* b2 = nextP(cx); const float* w2 = nextP(cx);
  const float* wn = (Cin != L.C) ? nextP(cx) : nullptr;

  do_bnrelu(cx, xin, L, Cin, g1, b1, L.tbf);
  do_conv(cx, L.tbf, w1, 27, Cin, L.C, L.y1, L.mask, MODE_NORM, L.S, L.S);
  do_bnrelu(cx, L.y1, L, L.C, g2, b2, L.tbf);
  do_conv(cx, L.tbf, w2, 27, L.C, L.C, L.y2, L.mask, MODE_NORM, L.S, L.S);

  const float* scp = xin;
  if (wn) {
    long n = L.V * Cin;
    k_tobf16<<<(unsigned)((n + 255) / 256), 256, 0, cx.st>>>(xin, L.catbf, n);
    do_conv(cx, L.catbf, wn, 1, Cin, L.C, L.sc, L.mask, MODE_NIN, L.S, L.S);
    scp = L.sc;
  }
  long n = L.V * L.C;
  k_add<<<(unsigned)((n + 255) / 256), 256, 0, cx.st>>>(scp, L.y2, xout, n);
}

static float* unet_rec(Ctx& cx, int l, float* xin)
{
  Lv& L = cx.lv[l];
  float* cur = xin;
  float* oth = (xin == L.xa) ? L.xb : L.xa;
  for (int i = 0; i < 2; ++i) {               // blocks1
    do_resblock(cx, L, cur, oth, L.C);
    float* t = cur; cur = oth; oth = t;
  }
  if (l == 6) return cur;

  // down
  const float* gd = nextP(cx); const float* bd = nextP(cx); const float* wd = nextP(cx);
  Lv& I = cx.lv[l + 1];
  do_bnrelu(cx, cur, L, L.C, gd, bd, L.tbf);
  do_conv(cx, L.tbf, wd, 8, L.C, I.C, I.xa, I.mask, MODE_DOWN, L.S, I.S);

  float* xi = unet_rec(cx, l + 1, I.xa);      // inner

  // up (conv_transpose onto parent active set)
  const float* gu = nextP(cx); const float* bu = nextP(cx); const float* wu = nextP(cx);
  do_bnrelu(cx, xi, I, I.C, gu, bu, I.tbf);
  do_conv(cx, I.tbf, wu, 8, I.C, L.C, L.xu, L.mask, MODE_UP, I.S, L.S);

  long cn = L.V * 2 * L.C;
  k_concat<<<(unsigned)((cn + 255) / 256), 256, 0, cx.st>>>(cur, L.xu, L.cat, L.V, L.C);

  // blocks2: first has NiN shortcut (2C -> C)
  do_resblock(cx, L, L.cat, cur, 2 * L.C);
  do_resblock(cx, L, cur, oth, L.C);
  return oth;
}

extern "C" void kernel_launch(void* const* d_in, const int* in_sizes, int n_in,
                              void* d_out, int out_size, void* d_ws, size_t ws_size,
                              hipStream_t stream)
{
  (void)n_in; (void)out_size; (void)ws_size;
  Ctx cx;
  cx.din = d_in; cx.st = stream;
  cx.base = (char*)d_ws; cx.off = 0;

  const int N = in_sizes[0] / 3;       // feats [N,3]
  const int S0 = 64;

  // Per-level buffers
  for (int l = 0; l < 7; ++l) {
    int S = 64 >> l; long V = (long)S * S * S; int C = 32 * (l + 1);
    Lv& L = cx.lv[l];
    L.S = S; L.V = V; L.C = C;
    L.xa   = (float*)ws_alloc(cx, (size_t)V * C * 4);
    L.xb   = (float*)ws_alloc(cx, (size_t)V * C * 4);
    L.y1   = (float*)ws_alloc(cx, (size_t)V * C * 4);
    L.y2   = (float*)ws_alloc(cx, (size_t)V * C * 4);
    L.xu   = (float*)ws_alloc(cx, (size_t)V * C * 4);
    L.sc   = L.xu;                     // xu dead after concat; reuse for NiN shortcut
    L.cat  = (float*)ws_alloc(cx, (size_t)V * 2 * C * 4);
    L.tbf  = (bf16*)ws_alloc(cx, (size_t)V * 2 * C * 2);
    L.catbf= (bf16*)ws_alloc(cx, (size_t)V * 2 * C * 2);
    L.mask = (float*)ws_alloc(cx, (size_t)V * 4);
    L.cnt  = (float*)ws_alloc(cx, 256);
  }
  cx.stats = (float*)ws_alloc(cx, 1024 * sizeof(float));
  cx.zrow  = (bf16*)ws_alloc(cx, 512 * sizeof(bf16));
  hipMemsetAsync(cx.zrow, 0, 512 * sizeof(bf16), stream);

  Lv& L0 = cx.lv[0];
  long V0 = L0.V;
  float* sums = (float*)ws_alloc(cx, (size_t)V0 * 3 * 4);
  float* cntg = (float*)ws_alloc(cx, (size_t)V0 * 4);
  float* grid = (float*)ws_alloc(cx, (size_t)V0 * 3 * 4);

  const float* feats = (const float*)d_in[0];
  const int*   coords = (const int*)d_in[1];
  const float* wstem = (const float*)d_in[2];

  // Input layer: average duplicate points into grid; build active mask
  hipMemsetAsync(sums, 0, (size_t)V0 * 3 * 4, stream);
  hipMemsetAsync(cntg, 0, (size_t)V0 * 4, stream);
  k_scatter<<<(unsigned)((N + 255) / 256), 256, 0, stream>>>(feats, coords, sums, cntg, N, S0);
  k_norm<<<(unsigned)((V0 + 255) / 256), 256, 0, stream>>>(sums, cntg, grid, L0.mask, V0);

  // Mask pyramid + active counts
  for (int l = 1; l < 7; ++l) {
    Lv& L = cx.lv[l];
    k_pool<<<(unsigned)((L.V + 255) / 256), 256, 0, stream>>>(cx.lv[l - 1].mask, L.mask, L.S);
  }
  for (int l = 0; l < 7; ++l) {
    Lv& L = cx.lv[l];
    hipMemsetAsync(L.cnt, 0, sizeof(float), stream);
    unsigned nb = (unsigned)((L.V + 255) / 256); if (nb > 256) nb = 256;
    k_count<<<nb, 256, 0, stream>>>(L.mask, L.cnt, L.V);
  }

  // Stem (Cin=3, scalar; ~0.7 GFLOP)
  k_stem<<<(unsigned)((V0 * 32 + 255) / 256), 256, 0, stream>>>(grid, L0.mask, wstem, L0.xa, S0);

  // U-Net (WMMA convs throughout)
  cx.pi = 3;
  float* xr = unet_rec(cx, 0, L0.xa);

  // Output head
  const float* go = nextP(cx); const float* bo = nextP(cx);
  const float* lw = nextP(cx); const float* lb = nextP(cx);
  do_bnrelu(cx, xr, L0, 32, go, bo, L0.tbf);
  k_final<<<(unsigned)(((long)N * 20 + 255) / 256), 256, 0, stream>>>(
      L0.tbf, coords, lw, lb, (float*)d_out, N, S0);
}
